// MaskedAutoencoder2DGeom_68367289418383
// MI455X (gfx1250) — matrix operand order repaired and checked by reading, not compiled, not run on hardware
//
#include <hip/hip_runtime.h>
#include <cmath>

#define B_      32
#define L_      576
#define PD_     768
#define E_      384
#define EH_     1536
#define NH_     6
#define DH_     64
#define DEPTH_  6
#define DE_     256
#define DEH_    1024
#define DNH_    8
#define DDH_    32
#define DDEPTH_ 4
#define KEEP_   288
#define GRID_   24

typedef __attribute__((ext_vector_type(16))) __bf16 v16bf;
typedef __attribute__((ext_vector_type(8)))  float  v8f;
typedef __attribute__((ext_vector_type(4)))  float  f32x4;
typedef __attribute__((ext_vector_type(4)))  unsigned int u32x4;
typedef __attribute__((ext_vector_type(2)))  unsigned int u32x2;

__device__ __forceinline__ unsigned short f2bf(float f) {
  unsigned int u = __builtin_bit_cast(unsigned int, f);
  u += 0x7fffu + ((u >> 16) & 1u);            // round-to-nearest-even
  return (unsigned short)(u >> 16);
}

__device__ __forceinline__ unsigned int pack2bf(float a, float b) {
  return (unsigned int)f2bf(a) | ((unsigned int)f2bf(b) << 16);
}

__device__ __forceinline__ float gelu_tanh(float x) {
  float x3 = x * x * x;
  return 0.5f * x * (1.0f + tanhf(0.7978845608028654f * (x + 0.044715f * x3)));
}

// ---------------------------------------------------------------------------
// Generic batched GEMM, bf16 WMMA, f32 accumulate, double-buffered LDS.
//   C[z] = epi( A[z] (MxK) * B[z] (KxN, or NxK if BT) + bias [+ R] )
// 256 threads = 8 waves; block tile 128x64, K-step 32; wave tile 32x32.
// Requirements (all satisfied by this model): K % 32 == 0, N % 4 == 0,
// lda/ldb and all base pointers 16B-aligned. M/N edges handled by index
// clamping (garbage flows only to never-stored output rows/cols).
// ---------------------------------------------------------------------------
#define BM 128
#define BN 64
#define BK 32
#define BKP 40   // padded LDS row stride (ushorts): 80B, 16B-aligned, conflict-free

__device__ __forceinline__ v16bf frag2(const unsigned short* p0, const unsigned short* p1) {
  union { u32x4 q[2]; v16bf v; } f;
  f.q[0] = *(const u32x4*)p0;
  f.q[1] = *(const u32x4*)p1;
  return f.v;
}

template<int EPI, bool BT>   // EPI: 0 = bias, 1 = bias+gelu, 2 = bias+residual
__global__ __launch_bounds__(256)
void gemm_wmma(const float* __restrict__ A, const float* __restrict__ Bm,
               const float* __restrict__ bias, const float* Rres, float* C,
               int M, int N, int K, int lda, int ldb, int ldc,
               long long sA, long long sB, long long sC)
{
  __shared__ __align__(16) unsigned short As[2][BM][BKP];
  __shared__ __align__(16) unsigned short Bs[2][BN][BKP];

  long long z = blockIdx.z;
  A  += z * sA;
  Bm += z * sB;
  C  += z * sC;
  const float* Rp = (EPI == 2) ? (Rres + z * sC) : nullptr;

  const int m0 = blockIdx.y * BM, n0 = blockIdx.x * BN;
  const int tid = threadIdx.x;
  const int wave = tid >> 5, lane = tid & 31;
  const int wm = wave >> 1, wn = wave & 1;    // waves: 4 (M) x 2 (N)
  const int ml = lane & 15, kg = lane >> 4;   // lane row/col, K half-group

  // ---- staging assignments (clamped row indices; unconditional b128 loads) ---
  const int qa  = tid & 7;     // float4 slot along K (A, and B when BT)
  const int rba = tid >> 3;    // row within a 32-row group
  int rowA[4];
  const float* pAc[4];
#pragma unroll
  for (int t = 0; t < 4; ++t) {
    int r  = t * 32 + rba;
    int gm = m0 + r;
    rowA[t] = r;
    pAc[t]  = A + (long long)(gm < M ? gm : 0) * lda + qa * 4;
  }
  int rB[2], kB[2], nq4 = (tid & 15) * 4;
  const float* pBc[2];
#pragma unroll
  for (int t = 0; t < 2; ++t) {
    if (BT) {
      int r  = t * 32 + rba;
      int gn = n0 + r;
      rB[t]  = r;
      pBc[t] = Bm + (long long)(gn < N ? gn : 0) * ldb + qa * 4;
    } else {
      int k  = t * 16 + (tid >> 4);
      int gn = n0 + nq4;
      kB[t]  = k;
      pBc[t] = Bm + (long long)k * ldb + (gn < N ? gn : 0);
    }
  }

  f32x4 ra[4], rb[2];
  auto loadT = [&]() {
#pragma unroll
    for (int t = 0; t < 4; ++t) { ra[t] = *(const f32x4*)pAc[t]; pAc[t] += BK; }
#pragma unroll
    for (int t = 0; t < 2; ++t) {
      rb[t] = *(const f32x4*)pBc[t];
      pBc[t] += BT ? (long long)BK : (long long)BK * ldb;
    }
  };
  auto storeT = [&](int b) {
#pragma unroll
    for (int t = 0; t < 4; ++t) {
      u32x2 d;
      d.x = pack2bf(ra[t].x, ra[t].y);
      d.y = pack2bf(ra[t].z, ra[t].w);
      *(u32x2*)&As[b][rowA[t]][qa * 4] = d;
    }
#pragma unroll
    for (int t = 0; t < 2; ++t) {
      if (BT) {
        u32x2 d;
        d.x = pack2bf(rb[t].x, rb[t].y);
        d.y = pack2bf(rb[t].z, rb[t].w);
        *(u32x2*)&Bs[b][rB[t]][qa * 4] = d;
      } else {
        Bs[b][nq4 + 0][kB[t]] = f2bf(rb[t].x);
        Bs[b][nq4 + 1][kB[t]] = f2bf(rb[t].y);
        Bs[b][nq4 + 2][kB[t]] = f2bf(rb[t].z);
        Bs[b][nq4 + 3][kB[t]] = f2bf(rb[t].w);
      }
    }
  };

  v8f acc00 = {}, acc01 = {}, acc10 = {}, acc11 = {};

  loadT();
  storeT(0);
  __syncthreads();
  int buf = 0;

  const int ra0 = wm * 32 + ml, ra1 = ra0 + 16;
  const int rb0 = wn * 32 + ml, rb1 = rb0 + 16;

  for (int k0 = 0; k0 < K; k0 += BK) {
    const bool notlast = (k0 + BK < K);
    if (notlast) loadT();                       // prefetch next tile (async vmem)

    const unsigned short (*Ab)[BKP] = As[buf];
    const unsigned short (*Bb)[BKP] = Bs[buf];
    // A frag: lane ml holds row M=ml; K = 8*kg+i (i<8), 16+8*kg+(i-8)
    v16bf a0 = frag2(&Ab[ra0][8 * kg], &Ab[ra0][16 + 8 * kg]);
    v16bf a1 = frag2(&Ab[ra1][8 * kg], &Ab[ra1][16 + 8 * kg]);
    // B frag: lane ml holds col N=ml; K = 8*kg+i (i<8), 16+8*kg+(i-8)
    v16bf b0 = frag2(&Bb[rb0][8 * kg], &Bb[rb0][16 + 8 * kg]);
    v16bf b1 = frag2(&Bb[rb1][8 * kg], &Bb[rb1][16 + 8 * kg]);

    acc00 = __builtin_amdgcn_wmma_f32_16x16x32_bf16(false, a0, false, b0, (short)0, acc00, false, false);
    acc01 = __builtin_amdgcn_wmma_f32_16x16x32_bf16(false, a0, false, b1, (short)0, acc01, false, false);
    acc10 = __builtin_amdgcn_wmma_f32_16x16x32_bf16(false, a1, false, b0, (short)0, acc10, false, false);
    acc11 = __builtin_amdgcn_wmma_f32_16x16x32_bf16(false, a1, false, b1, (short)0, acc11, false, false);

    if (notlast) {
      storeT(buf ^ 1);                          // fill other buffer
      __syncthreads();
      buf ^= 1;
    }
  }

  // D layout: lanes 0-15 VGPR r -> M=r ; lanes 16-31 VGPR r -> M=8+r ; N = lane&15
#pragma unroll
  for (int am = 0; am < 2; ++am) {
#pragma unroll
    for (int bn = 0; bn < 2; ++bn) {
      v8f acc = (am == 0) ? (bn == 0 ? acc00 : acc01) : (bn == 0 ? acc10 : acc11);
      int n = n0 + wn * 32 + bn * 16 + ml;
#pragma unroll
      for (int r = 0; r < 8; ++r) {
        int m = m0 + wm * 32 + am * 16 + kg * 8 + r;
        if (m < M && n < N) {
          float v = acc[r];
          if (bias) v += bias[n];
          if (EPI == 1) v = gelu_tanh(v);
          if (EPI == 2) v += Rp[(long long)m * ldc + n];
          C[(long long)m * ldc + n] = v;
        }
      }
    }
  }
}

// ---------------------------------------------------------------------------
// Elementwise / reduction kernels
// ---------------------------------------------------------------------------
__global__ void patchify_k(const float* __restrict__ imgs, float* __restrict__ target) {
  long long idx = (long long)blockIdx.x * 256 + threadIdx.x;
  if (idx >= (long long)B_ * L_ * PD_) return;
  int pd = idx % PD_;
  int l  = (idx / PD_) % L_;
  int b  = idx / ((long long)PD_ * L_);
  int c = pd % 3, t = pd / 3, pw = t % 16, ph = t / 16;
  int gh = l / GRID_, gw = l % GRID_;
  target[idx] = imgs[(((long long)b * 3 + c) * 384 + gh * 16 + ph) * 384 + gw * 16 + pw];
}

__global__ void addpos_k(float* __restrict__ x, const float* __restrict__ pos) {
  long long idx = (long long)blockIdx.x * 256 + threadIdx.x;
  if (idx >= (long long)B_ * L_ * E_) return;
  int e = idx % E_, l = (idx / E_) % L_;
  x[idx] += pos[l * E_ + e];
}

__global__ void sort_k(const float* __restrict__ noise, int* __restrict__ shuf,
                       int* __restrict__ rest) {
  int b = blockIdx.x;
  __shared__ float key[1024];
  __shared__ int   kid[1024];
  for (int i = threadIdx.x; i < 1024; i += 512) {
    key[i] = (i < L_) ? noise[b * L_ + i] : 3.0e38f;
    kid[i] = i;
  }
  __syncthreads();
  for (int k = 2; k <= 1024; k <<= 1) {
    for (int j = k >> 1; j > 0; j >>= 1) {
      for (int i = threadIdx.x; i < 1024; i += 512) {
        int ixj = i ^ j;
        if (ixj > i) {
          float a = key[i], c = key[ixj];
          int ia = kid[i], ic = kid[ixj];
          bool gt = (a > c) || (a == c && ia > ic);
          bool up = ((i & k) == 0);
          if (gt == up) { key[i] = c; key[ixj] = a; kid[i] = ic; kid[ixj] = ia; }
        }
      }
      __syncthreads();
    }
  }
  for (int i = threadIdx.x; i < L_; i += 512) {
    int v = kid[i];
    shuf[b * L_ + i] = v;
    rest[b * L_ + v] = i;
  }
}

__global__ void gather_k(const float* __restrict__ x, const int* __restrict__ ids,
                         float* __restrict__ xv) {
  long long idx = (long long)blockIdx.x * 256 + threadIdx.x;
  if (idx >= (long long)B_ * KEEP_ * E_) return;
  int e = idx % E_;
  int i = (idx / E_) % KEEP_;
  int b = idx / ((long long)E_ * KEEP_);
  xv[idx] = x[((long long)b * L_ + ids[b * L_ + i]) * E_ + e];
}

__global__ void mask_k(const int* __restrict__ rest, float* __restrict__ mask) {
  int idx = blockIdx.x * 256 + threadIdx.x;
  if (idx >= B_ * L_) return;
  mask[idx] = (rest[idx] >= KEEP_) ? 1.f : 0.f;
}

__global__ void layernorm_k(const float* __restrict__ x, const float* __restrict__ g,
                            const float* __restrict__ bt, float* __restrict__ y, int d) {
  int row = blockIdx.x;
  const float* xr = x + (long long)row * d;
  float* yr = y + (long long)row * d;
  __shared__ float s1[128], s2[128];
  float a = 0.f, b2 = 0.f;
  for (int i = threadIdx.x; i < d; i += 128) { float v = xr[i]; a += v; b2 += v * v; }
  s1[threadIdx.x] = a; s2[threadIdx.x] = b2;
  __syncthreads();
  for (int o = 64; o > 0; o >>= 1) {
    if (threadIdx.x < o) { s1[threadIdx.x] += s1[threadIdx.x + o]; s2[threadIdx.x] += s2[threadIdx.x + o]; }
    __syncthreads();
  }
  float mean = s1[0] / d;
  float var  = s2[0] / d - mean * mean;
  float rs = rsqrtf(var + 1e-5f);
  for (int i = threadIdx.x; i < d; i += 128) yr[i] = (xr[i] - mean) * rs * g[i] + bt[i];
}

// Fused scale + Manhattan-decay bias + softmax, in place on S (z = batch).
__global__ void softmax_k(float* __restrict__ S, const int* __restrict__ ids,
                          int Ntok, int ids_stride, float scale, float decay) {
  int bz = blockIdx.y, i = blockIdx.x, tid = threadIdx.x;
  float* row = S + ((long long)bz * Ntok + i) * Ntok;
  int pi = ids ? ids[bz * ids_stride + i] : i;
  int pih = pi / GRID_, piw = pi % GRID_;
  __shared__ float red[256];
  float mx = -3.0e38f;
  for (int j = tid; j < Ntok; j += 256) {
    int pj = ids ? ids[bz * ids_stride + j] : j;
    int qh = pj / GRID_, qw = pj % GRID_;
    int dd = (pih > qh ? pih - qh : qh - pih) + (piw > qw ? piw - qw : qw - piw);
    float v = row[j] * scale + decay * (float)dd;
    row[j] = v;
    mx = fmaxf(mx, v);
  }
  red[tid] = mx; __syncthreads();
  for (int o = 128; o > 0; o >>= 1) { if (tid < o) red[tid] = fmaxf(red[tid], red[tid + o]); __syncthreads(); }
  float m = red[0]; __syncthreads();
  float sum = 0.f;
  for (int j = tid; j < Ntok; j += 256) { float e = expf(row[j] - m); row[j] = e; sum += e; }
  red[tid] = sum; __syncthreads();
  for (int o = 128; o > 0; o >>= 1) { if (tid < o) red[tid] += red[tid + o]; __syncthreads(); }
  float inv = 1.f / red[0];
  for (int j = tid; j < Ntok; j += 256) row[j] *= inv;
}

__global__ void dec_assemble_k(const float* __restrict__ xd, const int* __restrict__ rest,
                               const float* __restrict__ mtok, const float* __restrict__ dpos,
                               float* __restrict__ xfull) {
  long long idx = (long long)blockIdx.x * 256 + threadIdx.x;
  if (idx >= (long long)B_ * L_ * DE_) return;
  int e = idx % DE_;
  int l = (idx / DE_) % L_;
  int b = idx / ((long long)DE_ * L_);
  int r = rest[b * L_ + l];
  float v = (r < KEEP_) ? xd[((long long)b * KEEP_ + r) * DE_ + e] : mtok[e];
  xfull[idx] = v + dpos[l * DE_ + e];
}

__global__ void zero2_k(float* acc) { acc[0] = 0.f; acc[1] = 0.f; }

__global__ void loss_k(const float* __restrict__ pred, const float* __restrict__ target,
                       const float* __restrict__ mask, float* acc) {
  int row = blockIdx.x;
  const float* p = pred + (long long)row * PD_;
  const float* t = target + (long long)row * PD_;
  __shared__ float red[256];
  float s = 0.f;
  for (int i = threadIdx.x; i < PD_; i += 256) { float d = p[i] - t[i]; s += d * d; }
  red[threadIdx.x] = s; __syncthreads();
  for (int o = 128; o > 0; o >>= 1) { if (threadIdx.x < o) red[threadIdx.x] += red[threadIdx.x + o]; __syncthreads(); }
  if (threadIdx.x == 0) {
    float mk = mask[row];
    atomicAdd(&acc[0], (red[0] / PD_) * mk);
    atomicAdd(&acc[1], mk);
  }
}

__global__ void finalize_k(const float* acc, float* loss) {
  loss[0] = acc[0] / (acc[1] + 1e-8f);
}

// ---------------------------------------------------------------------------
// Host orchestration
// ---------------------------------------------------------------------------
static void* bump(char** p, size_t bytes) {
  void* r = *p;
  *p += (bytes + 255) & ~(size_t)255;
  return r;
}

static void run_gemm(hipStream_t s, int epi, bool bt,
                     const float* A, const float* Bm, const float* bias,
                     const float* R, float* C, int M, int N, int K,
                     int lda, int ldb, int ldc,
                     long long sA, long long sB, long long sC, int Z)
{
  dim3 g((N + BN - 1) / BN, (M + BM - 1) / BM, Z);
  dim3 b(256);
  if (bt)            gemm_wmma<0, true ><<<g, b, 0, s>>>(A, Bm, bias, R, C, M, N, K, lda, ldb, ldc, sA, sB, sC);
  else if (epi == 0) gemm_wmma<0, false><<<g, b, 0, s>>>(A, Bm, bias, R, C, M, N, K, lda, ldb, ldc, sA, sB, sC);
  else if (epi == 1) gemm_wmma<1, false><<<g, b, 0, s>>>(A, Bm, bias, R, C, M, N, K, lda, ldb, ldc, sA, sB, sC);
  else               gemm_wmma<2, false><<<g, b, 0, s>>>(A, Bm, bias, R, C, M, N, K, lda, ldb, ldc, sA, sB, sC);
}

extern "C" void kernel_launch(void* const* d_in, const int* in_sizes, int n_in,
                              void* d_out, int out_size, void* d_ws, size_t ws_size,
                              hipStream_t stream)
{
  (void)in_sizes; (void)n_in; (void)out_size; (void)ws_size;
  // setup_inputs order with JAX pytree dict-key sorting inside each param stack:
  // stack keys sorted: W1,W2,Wo,Wqkv,b1,b2,bo,bqkv,ln1_b,ln1_g,ln2_b,ln2_g
  const float* imgs      = (const float*)d_in[0];
  const float* noise     = (const float*)d_in[1];
  const float* w_pe      = (const float*)d_in[2];
  const float* b_pe      = (const float*)d_in[3];
  const float* pos_embed = (const float*)d_in[4];
  const float* eW1   = (const float*)d_in[5];
  const float* eW2   = (const float*)d_in[6];
  const float* eWo   = (const float*)d_in[7];
  const float* eWqkv = (const float*)d_in[8];
  const float* eb1   = (const float*)d_in[9];
  const float* eb2   = (const float*)d_in[10];
  const float* ebo   = (const float*)d_in[11];
  const float* ebqkv = (const float*)d_in[12];
  const float* eln1b = (const float*)d_in[13];
  const float* eln1g = (const float*)d_in[14];
  const float* eln2b = (const float*)d_in[15];
  const float* eln2g = (const float*)d_in[16];
  const float* dW1   = (const float*)d_in[17];
  const float* dW2   = (const float*)d_in[18];
  const float* dWo   = (const float*)d_in[19];
  const float* dWqkv = (const float*)d_in[20];
  const float* db1   = (const float*)d_in[21];
  const float* db2   = (const float*)d_in[22];
  const float* dbo   = (const float*)d_in[23];
  const float* dbqkv = (const float*)d_in[24];
  const float* dln1b = (const float*)d_in[25];
  const float* dln1g = (const float*)d_in[26];
  const float* dln2b = (const float*)d_in[27];
  const float* dln2g = (const float*)d_in[28];
  const float* mask_token = (const float*)d_in[29];
  const float* dec_w  = (const float*)d_in[30];
  const float* dec_b  = (const float*)d_in[31];
  const float* dec_pos = (const float*)d_in[32];
  const float* pred_w = (const float*)d_in[33];
  const float* pred_b = (const float*)d_in[34];

  float* out      = (float*)d_out;
  float* out_loss = out;
  float* out_pred = out + 1;
  float* out_mask = out + 1 + (size_t)B_ * L_ * PD_;

  char* wp = (char*)d_ws;
  float* target = (float*)bump(&wp, sizeof(float) * (size_t)B_ * L_ * PD_);
  float* xtok   = (float*)bump(&wp, sizeof(float) * (size_t)B_ * L_ * E_);
  int*   shuf   = (int*)  bump(&wp, sizeof(int) * B_ * L_);
  int*   rest   = (int*)  bump(&wp, sizeof(int) * B_ * L_);
  float* xe     = (float*)bump(&wp, sizeof(float) * (size_t)B_ * KEEP_ * E_);
  float* hbuf   = (float*)bump(&wp, sizeof(float) * (size_t)B_ * L_ * DE_);     // >= B*KEEP*E
  float* qkv    = (float*)bump(&wp, sizeof(float) * (size_t)B_ * L_ * 3 * DE_); // >= B*KEEP*3E
  float* Sbuf   = (float*)bump(&wp, sizeof(float) * (size_t)B_ * L_ * L_);      // >= B*KEEP*KEEP
  float* attno  = (float*)bump(&wp, sizeof(float) * (size_t)B_ * L_ * DE_);     // >= B*KEEP*E
  float* mlp    = (float*)bump(&wp, sizeof(float) * (size_t)B_ * L_ * DEH_);    // >= B*KEEP*EH
  float* xd     = (float*)bump(&wp, sizeof(float) * (size_t)B_ * KEEP_ * DE_);
  float* xfull  = (float*)bump(&wp, sizeof(float) * (size_t)B_ * L_ * DE_);
  float* acc    = (float*)bump(&wp, sizeof(float) * 2);

  float decE[NH_], decD[DNH_];
  for (int h = 0; h < NH_;  ++h) decE[h] = logf(1.0f - exp2f(-5.0f - (float)h));
  for (int h = 0; h < DNH_; ++h) decD[h] = logf(1.0f - exp2f(-5.0f - (float)h));
  const float scE = 0.125f;                 // 1/sqrt(64)
  const float scD = 0.17677669529663687f;   // 1/sqrt(32)

  // ---- patchify + patch embed + pos embed ----
  {
    long long n = (long long)B_ * L_ * PD_;
    patchify_k<<<dim3((unsigned)((n + 255) / 256)), 256, 0, stream>>>(imgs, target);
  }
  run_gemm(stream, 0, false, target, w_pe, b_pe, nullptr, xtok,
           B_ * L_, E_, PD_, PD_, E_, E_, 0, 0, 0, 1);
  {
    long long n = (long long)B_ * L_ * E_;
    addpos_k<<<dim3((unsigned)((n + 255) / 256)), 256, 0, stream>>>(xtok, pos_embed);
  }

  // ---- masking ----
  sort_k<<<dim3(B_), 512, 0, stream>>>(noise, shuf, rest);
  {
    long long n = (long long)B_ * KEEP_ * E_;
    gather_k<<<dim3((unsigned)((n + 255) / 256)), 256, 0, stream>>>(xtok, shuf, xe);
  }
  mask_k<<<dim3((B_ * L_ + 255) / 256), 256, 0, stream>>>(rest, out_mask);

  // ---- encoder stack ----
  const int ME = B_ * KEEP_;
  for (int l = 0; l < DEPTH_; ++l) {
    layernorm_k<<<dim3(ME), 128, 0, stream>>>(xe, eln1g + l * E_, eln1b + l * E_, hbuf, E_);
    run_gemm(stream, 0, false, hbuf, eWqkv + (size_t)l * E_ * 3 * E_, ebqkv + l * 3 * E_,
             nullptr, qkv, ME, 3 * E_, E_, E_, 3 * E_, 3 * E_, 0, 0, 0, 1);
    for (int h = 0; h < NH_; ++h) {
      run_gemm(stream, 0, true,
               qkv + h * DH_, qkv + E_ + h * DH_, nullptr, nullptr, Sbuf,
               KEEP_, KEEP_, DH_, 3 * E_, 3 * E_, KEEP_,
               (long long)KEEP_ * 3 * E_, (long long)KEEP_ * 3 * E_, (long long)KEEP_ * KEEP_, B_);
      softmax_k<<<dim3(KEEP_, B_), 256, 0, stream>>>(Sbuf, shuf, KEEP_, L_, scE, decE[h]);
      run_gemm(stream, 0, false,
               Sbuf, qkv + 2 * E_ + h * DH_, nullptr, nullptr, attno + h * DH_,
               KEEP_, DH_, KEEP_, KEEP_, 3 * E_, E_,
               (long long)KEEP_ * KEEP_, (long long)KEEP_ * 3 * E_, (long long)KEEP_ * E_, B_);
    }
    run_gemm(stream, 2, false, attno, eWo + (size_t)l * E_ * E_, ebo + l * E_,
             xe, xe, ME, E_, E_, E_, E_, E_, 0, 0, 0, 1);
    layernorm_k<<<dim3(ME), 128, 0, stream>>>(xe, eln2g + l * E_, eln2b + l * E_, hbuf, E_);
    run_gemm(stream, 1, false, hbuf, eW1 + (size_t)l * E_ * EH_, eb1 + l * EH_,
             nullptr, mlp, ME, EH_, E_, E_, EH_, EH_, 0, 0, 0, 1);
    run_gemm(stream, 2, false, mlp, eW2 + (size_t)l * EH_ * E_, eb2 + l * E_,
             xe, xe, ME, E_, EH_, EH_, E_, E_, 0, 0, 0, 1);
  }

  // ---- decoder embed + assemble ----
  run_gemm(stream, 0, false, xe, dec_w, dec_b, nullptr, xd,
           ME, DE_, E_, E_, DE_, DE_, 0, 0, 0, 1);
  {
    long long n = (long long)B_ * L_ * DE_;
    dec_assemble_k<<<dim3((unsigned)((n + 255) / 256)), 256, 0, stream>>>(xd, rest, mask_token, dec_pos, xfull);
  }

  // ---- decoder stack ----
  const int MD = B_ * L_;
  for (int l = 0; l < DDEPTH_; ++l) {
    layernorm_k<<<dim3(MD), 128, 0, stream>>>(xfull, dln1g + l * DE_, dln1b + l * DE_, hbuf, DE_);
    run_gemm(stream, 0, false, hbuf, dWqkv + (size_t)l * DE_ * 3 * DE_, dbqkv + l * 3 * DE_,
             nullptr, qkv, MD, 3 * DE_, DE_, DE_, 3 * DE_, 3 * DE_, 0, 0, 0, 1);
    for (int h = 0; h < DNH_; ++h) {
      run_gemm(stream, 0, true,
               qkv + h * DDH_, qkv + DE_ + h * DDH_, nullptr, nullptr, Sbuf,
               L_, L_, DDH_, 3 * DE_, 3 * DE_, L_,
               (long long)L_ * 3 * DE_, (long long)L_ * 3 * DE_, (long long)L_ * L_, B_);
      softmax_k<<<dim3(L_, B_), 256, 0, stream>>>(Sbuf, nullptr, L_, 0, scD, decD[h]);
      run_gemm(stream, 0, false,
               Sbuf, qkv + 2 * DE_ + h * DDH_, nullptr, nullptr, attno + h * DDH_,
               L_, DDH_, L_, L_, 3 * DE_, DE_,
               (long long)L_ * L_, (long long)L_ * 3 * DE_, (long long)L_ * DE_, B_);
    }
    run_gemm(stream, 2, false, attno, dWo + (size_t)l * DE_ * DE_, dbo + l * DE_,
             xfull, xfull, MD, DE_, DE_, DE_, DE_, DE_, 0, 0, 0, 1);
    layernorm_k<<<dim3(MD), 128, 0, stream>>>(xfull, dln2g + l * DE_, dln2b + l * DE_, hbuf, DE_);
    run_gemm(stream, 1, false, hbuf, dW1 + (size_t)l * DE_ * DEH_, db1 + l * DEH_,
             nullptr, mlp, MD, DEH_, DE_, DE_, DEH_, DEH_, 0, 0, 0, 1);
    run_gemm(stream, 2, false, mlp, dW2 + (size_t)l * DEH_ * DE_, db2 + l * DE_,
             xfull, xfull, MD, DE_, DEH_, DEH_, DE_, DE_, 0, 0, 0, 1);
  }

  // ---- prediction head + loss ----
  run_gemm(stream, 0, false, xfull, pred_w, pred_b, nullptr, out_pred,
           MD, PD_, DE_, DE_, PD_, PD_, 0, 0, 0, 1);
  zero2_k<<<1, 1, 0, stream>>>(acc);
  loss_k<<<dim3(B_ * L_), 256, 0, stream>>>(out_pred, target, out_mask, acc);
  finalize_k<<<1, 1, 0, stream>>>(acc, out_loss);
}
